// MoE_22454089023919
// MI455X (gfx1250) — compile-verified, matrix operand order checked by reading, not run on hardware
//
#include <hip/hip_runtime.h>

#define T_TOK 8192
#define DIMX  2048
#define HIDX  1408
#define NE    8

typedef __bf16 bf16_t;
typedef __bf16 bf16x8  __attribute__((ext_vector_type(8)));
typedef __bf16 v16bf   __attribute__((ext_vector_type(16)));
typedef float  v8f     __attribute__((ext_vector_type(8)));
typedef int    v4i     __attribute__((ext_vector_type(4)));

// Address-space-qualified int4 pointers for the async-LDS builtin
// (param types per clang diagnostic: AS1 int4* src, AS3 int4* dst).
typedef __attribute__((address_space(1))) v4i as1_v4i;
typedef __attribute__((address_space(3))) v4i as3_v4i;

// ---------------------------------------------------------------------------
// CDNA5 async global->LDS copy (GLOBAL_LOAD_ASYNC_TO_LDS_B128, ASYNCcnt).
// Guarded: falls back to a synchronous copy if the builtin is unavailable.
// ---------------------------------------------------------------------------
#if defined(__has_builtin)
#if __has_builtin(__builtin_amdgcn_global_load_async_to_lds_b128) && \
    __has_builtin(__builtin_amdgcn_s_wait_asynccnt)
#define HAVE_ASYNC_LDS 1
#endif
#endif
#ifndef HAVE_ASYNC_LDS
#define HAVE_ASYNC_LDS 0
#endif

__device__ __forceinline__ void cp16_async(const bf16_t* g, bf16_t* l) {
#if HAVE_ASYNC_LDS
    // generic->LDS: hardware maps generic LDS addresses by discarding upper
    // bits, so the integer truncation to a 32-bit AS(3) pointer is faithful.
    __builtin_amdgcn_global_load_async_to_lds_b128(
        (as1_v4i*)(uintptr_t)g,
        (as3_v4i*)(uintptr_t)l,
        0, 0);
#else
    *(bf16x8*)l = *(const bf16x8*)g;
#endif
}

__device__ __forceinline__ void wait_async_lds() {
#if HAVE_ASYNC_LDS
    __builtin_amdgcn_s_wait_asynccnt(0);
#endif
}

// ---------------------------------------------------------------------------
// Fragment loader: 16x32 bf16 tile in LDS, row-major, stride 32 elements.
// CDNA5 16-bit A/B layout: lane<16 -> row=lane, K {0..7, 16..23};
//                          lane>=16 -> row=lane-16, K {8..15, 24..31}.
// Two ds_load_b128 per fragment.
// ---------------------------------------------------------------------------
__device__ __forceinline__ v16bf ld_frag(const bf16_t* base, int lane) {
    const int lr = lane & 15;
    const int hb = (lane >> 4) << 3;       // 0 or 8
    bf16x8 lo = *(const bf16x8*)(base + lr * 32 + hb);
    bf16x8 hi = *(const bf16x8*)(base + lr * 32 + 16 + hb);
    return __builtin_shufflevector(lo, hi, 0, 1, 2, 3, 4, 5, 6, 7,
                                   8, 9, 10, 11, 12, 13, 14, 15);
}

__device__ __forceinline__ v8f zero8() {
    v8f z = {0.f, 0.f, 0.f, 0.f, 0.f, 0.f, 0.f, 0.f};
    return z;
}

// ---------------------------------------------------------------------------
// Router: 1 wave per token. Lane-parallel dot products over DIM, wave32
// xor-reduction, softmax + top-2 on lane 0, build per-expert token lists.
// ---------------------------------------------------------------------------
__global__ void moe_router(const float* __restrict__ x,
                           const float* __restrict__ gw,
                           float* __restrict__ combine,
                           int* __restrict__ counts,
                           int* __restrict__ lists) {
    const int wid  = threadIdx.x >> 5;
    const int lane = threadIdx.x & 31;
    const int t = blockIdx.x * 8 + wid;
    if (t >= T_TOK) return;

    const float* xr = x + (size_t)t * DIMX;
    float acc[NE];
#pragma unroll
    for (int e = 0; e < NE; ++e) acc[e] = 0.f;

    for (int d = lane; d < DIMX; d += 32) {
        const float xv = xr[d];
#pragma unroll
        for (int e = 0; e < NE; ++e)
            acc[e] = __builtin_fmaf(xv, gw[e * DIMX + d], acc[e]);
    }
#pragma unroll
    for (int e = 0; e < NE; ++e) {
#pragma unroll
        for (int m = 16; m >= 1; m >>= 1)
            acc[e] += __shfl_xor(acc[e], m, 32);
    }
    if (lane == 0) {
        float mx = acc[0];
#pragma unroll
        for (int e = 1; e < NE; ++e) mx = fmaxf(mx, acc[e]);
        float p[NE], s = 0.f;
#pragma unroll
        for (int e = 0; e < NE; ++e) { p[e] = __expf(acc[e] - mx); s += p[e]; }
        const float inv = 1.f / s;
#pragma unroll
        for (int e = 0; e < NE; ++e) p[e] *= inv;

        int i0 = 0; float v0 = p[0];
#pragma unroll
        for (int e = 1; e < NE; ++e) if (p[e] > v0) { v0 = p[e]; i0 = e; }
        int i1 = -1; float v1 = -1.f;
#pragma unroll
        for (int e = 0; e < NE; ++e)
            if (e != i0 && p[e] > v1) { v1 = p[e]; i1 = e; }

#pragma unroll
        for (int e = 0; e < NE; ++e) combine[(size_t)t * NE + e] = 0.f;
        combine[(size_t)t * NE + i0] = v0;
        combine[(size_t)t * NE + i1] = v1;

        int s0 = atomicAdd(&counts[i0], 1); lists[i0 * T_TOK + s0] = t;
        int s1 = atomicAdd(&counts[i1], 1); lists[i1 * T_TOK + s1] = t;
    }
}

__global__ void init_counts(int* counts) {
    if (threadIdx.x < NE) counts[threadIdx.x] = 0;
}

// ---------------------------------------------------------------------------
// Elementwise fp32 -> bf16 cast (x activations), 4 per thread.
// ---------------------------------------------------------------------------
__global__ void cast_f32_bf16(const float* __restrict__ in,
                              bf16_t* __restrict__ out, int n) {
    int i = (blockIdx.x * blockDim.x + threadIdx.x) * 4;
    if (i < n) {
        float4 v = *(const float4*)(in + i);
        out[i + 0] = (bf16_t)v.x;
        out[i + 1] = (bf16_t)v.y;
        out[i + 2] = (bf16_t)v.z;
        out[i + 3] = (bf16_t)v.w;
    }
}

// ---------------------------------------------------------------------------
// Tiled transpose + cast: in fp32 [R,C] -> out bf16 [C,R] (batched on z).
// Puts K contiguous for WMMA B-fragments; bf16 weights (~155MB) fit in L2.
// ---------------------------------------------------------------------------
__global__ void transpose_cast(const float* __restrict__ in,
                               bf16_t* __restrict__ out, int R, int C) {
    __shared__ float tile[32][33];
    const int bx = blockIdx.x * 32;                 // C offset
    const int by = blockIdx.y * 32;                 // R offset
    const size_t batch = (size_t)blockIdx.z * R * C;
    const int tx = threadIdx.x & 31, ty = threadIdx.x >> 5;   // 32 x 8
#pragma unroll
    for (int j = 0; j < 32; j += 8)
        tile[ty + j][tx] = in[batch + (size_t)(by + ty + j) * C + bx + tx];
    __syncthreads();
#pragma unroll
    for (int j = 0; j < 32; j += 8)
        out[batch + (size_t)(bx + ty + j) * R + by + tx] = (bf16_t)tile[tx][ty + j];
}

// ---------------------------------------------------------------------------
// FFN phase 1: h = silu(Xg @ W1) * (Xg @ W3), gathered rows, bf16 out.
// Block tile 128(M) x 64(N), 8 waves of 32x32, K-step 32, double-buffered
// LDS filled by the async DMA path (overlaps with WMMA); 1 barrier / K-step.
// launch_bounds(256,1): full VGPR budget, no spills; latency hidden by DMA.
// ---------------------------------------------------------------------------
__global__ void __launch_bounds__(256, 1)
moe_ffn1(const bf16_t* __restrict__ xbf,
         const bf16_t* __restrict__ w1t,   // [HID, DIM]
         const bf16_t* __restrict__ w3t,   // [HID, DIM]
         bf16_t* __restrict__ hbuf,        // [T, HID]
         const int* __restrict__ list,
         const int* __restrict__ countp) {
    __shared__ __align__(16) bf16_t AsB[2][128 * 32];
    __shared__ __align__(16) bf16_t B1B[2][64 * 32];
    __shared__ __align__(16) bf16_t B3B[2][64 * 32];
    __shared__ int tok[128];

    const int rows = countp ? countp[0] : T_TOK;
    const int bm = blockIdx.x;
    if (bm * 128 >= rows) return;
    const int n0 = blockIdx.y * 64;

    const int tid  = threadIdx.x;
    const int lane = tid & 31, wid = tid >> 5;
    const int wm = wid & 3, wn = wid >> 2;

    if (tid < 128) {
        const int g = bm * 128 + tid;
        int t;
        if (list) t = (g < rows) ? list[g] : list[rows - 1];
        else      t = (g < rows) ? g : 0;
        tok[tid] = t;
    }
    __syncthreads();

    const int arow = tid >> 1, ahalf = (tid & 1) * 16;   // 128 rows x 32
    const int brow = tid >> 2, bq = (tid & 3) * 8;       // 64 rows x 32
    const size_t abase = (size_t)tok[arow] * DIMX + ahalf;
    const size_t wbase = (size_t)(n0 + brow) * DIMX + bq;

    auto stage = [&](int k0, int b) {
        const bf16_t* asrc = xbf + abase + k0;
        cp16_async(asrc,     AsB[b] + arow * 32 + ahalf);
        cp16_async(asrc + 8, AsB[b] + arow * 32 + ahalf + 8);
        cp16_async(w1t + wbase + k0, B1B[b] + brow * 32 + bq);
        cp16_async(w3t + wbase + k0, B3B[b] + brow * 32 + bq);
    };

    v8f acc1[2][2], acc3[2][2];
#pragma unroll
    for (int i = 0; i < 2; ++i)
#pragma unroll
        for (int j = 0; j < 2; ++j) { acc1[i][j] = zero8(); acc3[i][j] = zero8(); }

    stage(0, 0);

    const int kSteps = DIMX / 32;
#pragma unroll 1
    for (int ks = 0; ks < kSteps; ++ks) {
        wait_async_lds();      // our async copies into buffer `cur` complete
        __syncthreads();       // all waves' copies done / prior reads drained
        const int cur = ks & 1;
        if (ks + 1 < kSteps) stage((ks + 1) * 32, cur ^ 1);

        const bf16_t* As = AsB[cur];
        const bf16_t* B1 = B1B[cur];
        const bf16_t* B3 = B3B[cur];

        v16bf a0  = ld_frag(As + (wm * 32 + 0)  * 32, lane);
        v16bf a1  = ld_frag(As + (wm * 32 + 16) * 32, lane);
        v16bf b10 = ld_frag(B1 + (wn * 32 + 0)  * 32, lane);
        v16bf b11 = ld_frag(B1 + (wn * 32 + 16) * 32, lane);
        v16bf b30 = ld_frag(B3 + (wn * 32 + 0)  * 32, lane);
        v16bf b31 = ld_frag(B3 + (wn * 32 + 16) * 32, lane);

        acc1[0][0] = __builtin_amdgcn_wmma_f32_16x16x32_bf16(false, a0, false, b10, (short)0, acc1[0][0], false, false);
        acc1[0][1] = __builtin_amdgcn_wmma_f32_16x16x32_bf16(false, a0, false, b11, (short)0, acc1[0][1], false, false);
        acc1[1][0] = __builtin_amdgcn_wmma_f32_16x16x32_bf16(false, a1, false, b10, (short)0, acc1[1][0], false, false);
        acc1[1][1] = __builtin_amdgcn_wmma_f32_16x16x32_bf16(false, a1, false, b11, (short)0, acc1[1][1], false, false);
        acc3[0][0] = __builtin_amdgcn_wmma_f32_16x16x32_bf16(false, a0, false, b30, (short)0, acc3[0][0], false, false);
        acc3[0][1] = __builtin_amdgcn_wmma_f32_16x16x32_bf16(false, a0, false, b31, (short)0, acc3[0][1], false, false);
        acc3[1][0] = __builtin_amdgcn_wmma_f32_16x16x32_bf16(false, a1, false, b30, (short)0, acc3[1][0], false, false);
        acc3[1][1] = __builtin_amdgcn_wmma_f32_16x16x32_bf16(false, a1, false, b31, (short)0, acc3[1][1], false, false);
    }

    // epilogue: fused SiLU * up, bf16 store (C layout: lane<16 M=j, lane>=16 M=8+j)
    const int lr = lane & 15, hb = lane >> 4;
#pragma unroll
    for (int mi = 0; mi < 2; ++mi) {
#pragma unroll
        for (int ni = 0; ni < 2; ++ni) {
            const v8f a = acc1[mi][ni], b = acc3[mi][ni];
            const int col = n0 + wn * 32 + ni * 16 + lr;
#pragma unroll
            for (int j = 0; j < 8; ++j) {
                const int g = bm * 128 + wm * 32 + mi * 16 + hb * 8 + j;
                if (g < rows) {
                    const float av = a[j];
                    const float hv = (av / (1.f + __expf(-av))) * b[j];
                    hbuf[(size_t)g * HIDX + col] = (bf16_t)hv;
                }
            }
        }
    }
}

// ---------------------------------------------------------------------------
// FFN phase 2: Y = H @ W2, scaled scatter-add into out (plain store for the
// shared expert, which runs first and initializes out). w2t is [DIM, HID].
// ---------------------------------------------------------------------------
__global__ void __launch_bounds__(256, 1)
moe_ffn2(const bf16_t* __restrict__ hbuf,
         const bf16_t* __restrict__ w2t,
         float* __restrict__ out,
         const int* __restrict__ list,
         const int* __restrict__ countp,
         const float* __restrict__ combine,
         int expert) {
    __shared__ __align__(16) bf16_t AsB[2][128 * 32];
    __shared__ __align__(16) bf16_t BtB[2][64 * 32];
    __shared__ int tok[128];
    __shared__ float rs[128];

    const int rows = countp ? countp[0] : T_TOK;
    const int bm = blockIdx.x;
    if (bm * 128 >= rows) return;
    const int n0 = blockIdx.y * 64;

    const int tid  = threadIdx.x;
    const int lane = tid & 31, wid = tid >> 5;
    const int wm = wid & 3, wn = wid >> 2;

    if (tid < 128) {
        const int g = bm * 128 + tid;
        int t;
        if (list) t = (g < rows) ? list[g] : list[rows - 1];
        else      t = (g < rows) ? g : 0;
        tok[tid] = t;
        rs[tid] = (expert >= 0 && g < rows) ? combine[(size_t)t * NE + expert] : 1.f;
    }
    __syncthreads();

    const int arow = tid >> 1, ahalf = (tid & 1) * 16;
    const int brow = tid >> 2, bq = (tid & 3) * 8;
    const size_t abase = (size_t)(bm * 128 + arow) * HIDX + ahalf;
    const size_t wbase = (size_t)(n0 + brow) * HIDX + bq;

    auto stage = [&](int k0, int b) {
        const bf16_t* asrc = hbuf + abase + k0;
        cp16_async(asrc,     AsB[b] + arow * 32 + ahalf);
        cp16_async(asrc + 8, AsB[b] + arow * 32 + ahalf + 8);
        cp16_async(w2t + wbase + k0, BtB[b] + brow * 32 + bq);
    };

    v8f acc[2][2];
#pragma unroll
    for (int i = 0; i < 2; ++i)
#pragma unroll
        for (int j = 0; j < 2; ++j) acc[i][j] = zero8();

    stage(0, 0);

    const int kSteps = HIDX / 32;
#pragma unroll 1
    for (int ks = 0; ks < kSteps; ++ks) {
        wait_async_lds();
        __syncthreads();
        const int cur = ks & 1;
        if (ks + 1 < kSteps) stage((ks + 1) * 32, cur ^ 1);

        const bf16_t* As = AsB[cur];
        const bf16_t* Bt = BtB[cur];

        v16bf a0 = ld_frag(As + (wm * 32 + 0)  * 32, lane);
        v16bf a1 = ld_frag(As + (wm * 32 + 16) * 32, lane);
        v16bf b0 = ld_frag(Bt + (wn * 32 + 0)  * 32, lane);
        v16bf b1 = ld_frag(Bt + (wn * 32 + 16) * 32, lane);

        acc[0][0] = __builtin_amdgcn_wmma_f32_16x16x32_bf16(false, a0, false, b0, (short)0, acc[0][0], false, false);
        acc[0][1] = __builtin_amdgcn_wmma_f32_16x16x32_bf16(false, a0, false, b1, (short)0, acc[0][1], false, false);
        acc[1][0] = __builtin_amdgcn_wmma_f32_16x16x32_bf16(false, a1, false, b0, (short)0, acc[1][0], false, false);
        acc[1][1] = __builtin_amdgcn_wmma_f32_16x16x32_bf16(false, a1, false, b1, (short)0, acc[1][1], false, false);
    }

    const int lr = lane & 15, hb = lane >> 4;
#pragma unroll
    for (int mi = 0; mi < 2; ++mi) {
#pragma unroll
        for (int ni = 0; ni < 2; ++ni) {
            const v8f a = acc[mi][ni];
            const int col = n0 + wn * 32 + ni * 16 + lr;
#pragma unroll
            for (int j = 0; j < 8; ++j) {
                const int lrow = wm * 32 + mi * 16 + hb * 8 + j;
                const int g = bm * 128 + lrow;
                if (g < rows) {
                    const int t = tok[lrow];
                    float* dst = out + (size_t)t * DIMX + col;
                    if (expert < 0) *dst = a[j];                        // shared expert init
                    else atomicAdd(dst, rs[lrow] * a[j]);               // routed accumulate
                }
            }
        }
    }
}

// ---------------------------------------------------------------------------
// Host launcher
// ---------------------------------------------------------------------------
static inline size_t align_up(size_t v) { return (v + 255) & ~(size_t)255; }

extern "C" void kernel_launch(void* const* d_in, const int* in_sizes, int n_in,
                              void* d_out, int out_size, void* d_ws, size_t ws_size,
                              hipStream_t stream) {
    const float* x      = (const float*)d_in[0];
    const float* gate_w = (const float*)d_in[1];
    const float* w1     = (const float*)d_in[2];
    const float* w2     = (const float*)d_in[3];
    const float* w3     = (const float*)d_in[4];
    const float* sw1    = (const float*)d_in[5];
    const float* sw2    = (const float*)d_in[6];
    const float* sw3    = (const float*)d_in[7];
    float* out = (float*)d_out;

    char* base = (char*)d_ws;
    size_t off = 0;
    auto take = [&](size_t bytes) { char* p = base + off; off += align_up(bytes); return p; };

    const size_t XN   = (size_t)T_TOK * DIMX;          // 16.7M
    const size_t WN   = (size_t)NE * DIMX * HIDX;      // 23.1M per tensor
    const size_t SWN  = (size_t)DIMX * HIDX;           // 2.88M

    bf16_t* xbf  = (bf16_t*)take(XN  * 2);
    bf16_t* w1t  = (bf16_t*)take(WN  * 2);             // [e][HID][DIM]
    bf16_t* w3t  = (bf16_t*)take(WN  * 2);             // [e][HID][DIM]
    bf16_t* w2t  = (bf16_t*)take(WN  * 2);             // [e][DIM][HID]
    bf16_t* sw1t = (bf16_t*)take(SWN * 2);
    bf16_t* sw3t = (bf16_t*)take(SWN * 2);
    bf16_t* sw2t = (bf16_t*)take(SWN * 2);
    float*  combine = (float*)take((size_t)T_TOK * NE * 4);
    int*    counts  = (int*)take(64 * 4);
    int*    lists   = (int*)take((size_t)NE * T_TOK * 4);
    bf16_t* hbuf    = (bf16_t*)take((size_t)T_TOK * HIDX * 2);

    init_counts<<<1, 64, 0, stream>>>(counts);
    cast_f32_bf16<<<(int)((XN / 4 + 255) / 256), 256, 0, stream>>>(x, xbf, (int)XN);

    transpose_cast<<<dim3(HIDX / 32, DIMX / 32, NE), 256, 0, stream>>>(w1, w1t, DIMX, HIDX);
    transpose_cast<<<dim3(HIDX / 32, DIMX / 32, NE), 256, 0, stream>>>(w3, w3t, DIMX, HIDX);
    transpose_cast<<<dim3(DIMX / 32, HIDX / 32, NE), 256, 0, stream>>>(w2, w2t, HIDX, DIMX);
    transpose_cast<<<dim3(HIDX / 32, DIMX / 32, 1),  256, 0, stream>>>(sw1, sw1t, DIMX, HIDX);
    transpose_cast<<<dim3(HIDX / 32, DIMX / 32, 1),  256, 0, stream>>>(sw3, sw3t, DIMX, HIDX);
    transpose_cast<<<dim3(DIMX / 32, HIDX / 32, 1),  256, 0, stream>>>(sw2, sw2t, HIDX, DIMX);

    moe_router<<<T_TOK / 8, 256, 0, stream>>>(x, gate_w, combine, counts, lists);

    // shared expert: initializes out with plain stores
    moe_ffn1<<<dim3(T_TOK / 128, HIDX / 64), 256, 0, stream>>>(xbf, sw1t, sw3t, hbuf, nullptr, nullptr);
    moe_ffn2<<<dim3(T_TOK / 128, DIMX / 64), 256, 0, stream>>>(hbuf, sw2t, out, nullptr, nullptr, nullptr, -1);

    // routed experts: gathered rows, scaled atomic accumulate
    for (int e = 0; e < NE; ++e) {
        moe_ffn1<<<dim3(T_TOK / 128, HIDX / 64), 256, 0, stream>>>(
            xbf, w1t + (size_t)e * HIDX * DIMX, w3t + (size_t)e * HIDX * DIMX,
            hbuf, lists + (size_t)e * T_TOK, counts + e);
        moe_ffn2<<<dim3(T_TOK / 128, DIMX / 64), 256, 0, stream>>>(
            hbuf, w2t + (size_t)e * DIMX * HIDX, out,
            lists + (size_t)e * T_TOK, counts + e, combine, e);
    }
}
